// TinyNeRF_75866302316858
// MI455X (gfx1250) — matrix operand order repaired and checked by reading, not compiled, not run on hardware
//
#include <hip/hip_runtime.h>
#include <math.h>

typedef __attribute__((ext_vector_type(16))) _Float16 v16h;
typedef __attribute__((ext_vector_type(8)))  _Float16 v8h;
typedef __attribute__((ext_vector_type(8)))  float    v8f;

typedef __attribute__((address_space(3))) _Float16 lds_h;
typedef __attribute__((address_space(3))) const float lds_cf;

#define NPTS            1048576
#define NUM_UNITS       (NPTS / 16)                // 65536 16-point work units
#define WAVES_PER_BLOCK 12
#define BLOCK_THREADS   (WAVES_PER_BLOCK * 32)     // 384
#define GRID_BLOCKS     512

#define SA 136   // bufA stride (halves): holds <=128 cols. 272B row -> conflict-free b128
#define SB 168   // bufB stride (halves): holds <=160 cols. 336B row -> conflict-free b128
#define G1_STRIDE  72    // g1 weights: K padded 36 -> 64, stride 72
#define GW_STRIDE  136   // 128x128 layers: K=128, stride 136
#define C1_STRIDE  168   // c1: K padded 152 -> 160, stride 168

struct alignas(16) Smem {
    _Float16 g1t[128 * G1_STRIDE];   // [n][k] transposed, zero-padded k>=36
    _Float16 g2t[128 * GW_STRIDE];
    _Float16 g3t[128 * GW_STRIDE];
    _Float16 g4t[128 * GW_STRIDE];
    _Float16 ft [128 * GW_STRIDE];
    _Float16 c1t[64  * C1_STRIDE];   // zero-padded k>=152
    float gb[5][128];                // biases g1,g2,g3,g4,f
    float c1b[64];
    float c2w[64 * 3];
    float c2b[3];
    float dw[128];
    float db[1];
    _Float16 actA[WAVES_PER_BLOCK][16 * SA];   // ping buffer (<=128 cols)
    _Float16 actB[WAVES_PER_BLOCK][16 * SB];   // pong buffer (<=160 cols)
    _Float16 dirS[WAVES_PER_BLOCK][16 * 24];   // dir encoding stash
};
// LDS total ~302 KB <= 320 KB/WGP (CDNA5)

__device__ __forceinline__ void lds_fence() {
    asm volatile("s_wait_dscnt 0" ::: "memory");
}

// One dense layer: act(16xK f16, LDS, stride SIN) @ Wt -> actOut(16 x 16*NT f16, stride SOUT)
// Wt stored transposed [n][k] with row stride WS (halves).
// A fragments preloaded once; store/bias bases pinned in a VGPR (AS3, 32-bit) so all
// j/r indexing folds into 16-bit DS immediate offsets; j-loop unrolled x4 for DS/WMMA overlap.
template <int KT, int NT, bool RELU, int SIN, int SOUT, int WS>
__device__ __forceinline__ void layer(const _Float16* actIn, const _Float16* wt,
                                      const float* bias, _Float16* actOut, int lane)
{
    lds_fence();                       // prior cross-lane LDS writes must land
    const int m16  = lane & 15;
    const int half = lane >> 4;

    // Preload all A fragments for this 16-row tile (reused across all NT output tiles).
    v16h afrag[KT];
    #pragma unroll
    for (int kc = 0; kc < KT; ++kc) {
        // half 0 -> K {kb..kb+7, kb+16..kb+23}; half 1 -> +8  (ISA A-layout)
        const _Float16* pa = actIn + m16 * SIN + kc * 32 + half * 8;
        v8h alo = *(const v8h*)pa;
        v8h ahi = *(const v8h*)(pa + 16);
        #pragma unroll
        for (int i = 0; i < 8; ++i) { afrag[kc][i] = alo[i]; afrag[kc][i + 8] = ahi[i]; }
    }

    // Pinned 32-bit LDS base pointers: one VGPR each, constant indexing folds into
    // the DS instruction's immediate offset (max 7*16 + 7*SOUT halves < 64KB).
    lds_h* outp = (lds_h*)(actOut + half * 8 * SOUT + m16);
    asm("" : "+v"(outp));
    lds_cf* bp = (lds_cf*)(bias + m16);
    asm("" : "+v"(bp));

    #pragma unroll 4
    for (int j = 0; j < NT; ++j) {
        v8f c = {};
        // Single base for B fragments of this tile; kc folds into immediates.
        const _Float16* pb = wt + (j * 16 + m16) * WS + half * 8;
        #pragma unroll
        for (int kc = 0; kc < KT; ++kc) {
            v8h blo = *(const v8h*)(pb + kc * 32);
            v8h bhi = *(const v8h*)(pb + kc * 32 + 16);
            v16h b;
            #pragma unroll
            for (int i = 0; i < 8; ++i) { b[i] = blo[i]; b[i + 8] = bhi[i]; }
            c = __builtin_amdgcn_wmma_f32_16x16x32_f16(
                    false, afrag[kc], false, b, (short)0, c, false, false);
        }
        const float bv = bp[j * 16];
        #pragma unroll
        for (int r = 0; r < 8; ++r) {
            float v = c[r] + bv;
            if (RELU) v = fmaxf(v, 0.0f);
            // C element (m = r + 8*half, n = j*16 + m16)
            outp[j * 16 + r * SOUT] = (_Float16)v;
        }
    }
}

__global__ __launch_bounds__(BLOCK_THREADS)
__attribute__((amdgpu_num_vgpr(256)))
void tinynerf_wmma(const float* __restrict__ positions,
                   const float* __restrict__ directions,
                   const float* __restrict__ g1_w, const float* __restrict__ g1_b,
                   const float* __restrict__ g2_w, const float* __restrict__ g2_b,
                   const float* __restrict__ g3_w, const float* __restrict__ g3_b,
                   const float* __restrict__ g4_w, const float* __restrict__ g4_b,
                   const float* __restrict__ d_w,  const float* __restrict__ d_b,
                   const float* __restrict__ f_w,  const float* __restrict__ f_b,
                   const float* __restrict__ c1_w, const float* __restrict__ c1_b,
                   const float* __restrict__ c2_w, const float* __restrict__ c2_b,
                   float* __restrict__ out)
{
    __shared__ Smem sm;
    const int tid = threadIdx.x;

    // ---- Cooperative weight load (coalesced: fan-out index fastest) + transpose ----
    for (int e = tid; e < 36 * 128; e += BLOCK_THREADS) {
        int k = e >> 7, n = e & 127;
        sm.g1t[n * G1_STRIDE + k] = (_Float16)g1_w[e];
    }
    for (int e = tid; e < 128 * (G1_STRIDE - 36); e += BLOCK_THREADS) {
        int n = e / (G1_STRIDE - 36), k = 36 + e % (G1_STRIDE - 36);
        sm.g1t[n * G1_STRIDE + k] = (_Float16)0.0f;
    }
    for (int e = tid; e < 128 * 128; e += BLOCK_THREADS) {
        int k = e >> 7, n = e & 127;
        sm.g2t[n * GW_STRIDE + k] = (_Float16)g2_w[e];
        sm.g3t[n * GW_STRIDE + k] = (_Float16)g3_w[e];
        sm.g4t[n * GW_STRIDE + k] = (_Float16)g4_w[e];
        sm.ft [n * GW_STRIDE + k] = (_Float16)f_w [e];
    }
    for (int e = tid; e < 128 * (GW_STRIDE - 128); e += BLOCK_THREADS) {
        int n = e >> 3, k = 128 + (e & 7);
        _Float16 z = (_Float16)0.0f;
        sm.g2t[n * GW_STRIDE + k] = z;
        sm.g3t[n * GW_STRIDE + k] = z;
        sm.g4t[n * GW_STRIDE + k] = z;
        sm.ft [n * GW_STRIDE + k] = z;
    }
    for (int e = tid; e < 152 * 64; e += BLOCK_THREADS) {
        int k = e >> 6, n = e & 63;
        sm.c1t[n * C1_STRIDE + k] = (_Float16)c1_w[e];
    }
    for (int e = tid; e < 64 * (C1_STRIDE - 152); e += BLOCK_THREADS) {
        int n = e >> 4, k = 152 + (e & 15);
        sm.c1t[n * C1_STRIDE + k] = (_Float16)0.0f;
    }
    for (int i = tid; i < 128; i += BLOCK_THREADS) {
        sm.gb[0][i] = g1_b[i]; sm.gb[1][i] = g2_b[i]; sm.gb[2][i] = g3_b[i];
        sm.gb[3][i] = g4_b[i]; sm.gb[4][i] = f_b[i];
        sm.dw[i] = d_w[i];
    }
    for (int i = tid; i < 64; i += BLOCK_THREADS) sm.c1b[i] = c1_b[i];
    for (int i = tid; i < 192; i += BLOCK_THREADS) sm.c2w[i] = c2_w[i];
    if (tid < 3) sm.c2b[tid] = c2_b[tid];
    if (tid == 3) sm.db[0] = d_b[0];
    __syncthreads();

    const int wave = tid >> 5;
    const int lane = tid & 31;
    const int m    = lane & 15;
    const int half = lane >> 4;
    _Float16* bufA = sm.actA[wave];
    _Float16* bufB = sm.actB[wave];
    _Float16* dirS = sm.dirS[wave];
    const float PI = 3.14159265358979323846f;

    // Per-wave persistent loop over 16-point work units (wave-uniform -> EXEC all-ones)
    const int totalWaves = gridDim.x * WAVES_PER_BLOCK;
    const int waveG = blockIdx.x * WAVES_PER_BLOCK + wave;

    for (int wt = waveG; wt < NUM_UNITS; wt += totalWaves) {
        const int p = wt * 16 + m;

        // Prefetch next unit's inputs into cache (global_prefetch_b8)
        const int wtn = wt + totalWaves;
        if (wtn < NUM_UNITS) {
            const int pn = wtn * 16 + m;
            __builtin_prefetch(&positions[pn * 3], 0, 1);
            __builtin_prefetch(&directions[pn * 3], 0, 1);
        }

        // ---- positional encoding -> bufA rows (K = 36, padded to 64) ----
        // Branch-free: compute sin and cos, per-lane select (v_cndmask), no EXEC churn.
        float pv[3] = { positions[p * 3 + 0], positions[p * 3 + 1], positions[p * 3 + 2] };
        _Float16* encp = bufA + m * SA + half * 6;
        #pragma unroll
        for (int d = 0; d < 3; ++d) {
            float x = pv[d] * PI;
            float f = 1.0f;
            #pragma unroll
            for (int fr = 0; fr < 6; ++fr) {
                float ang = x * f;
                float sv = __sinf(ang), cv = __cosf(ang);
                float v = half ? cv : sv;
                encp[d * 12 + fr] = (_Float16)v;
                f *= 2.0f;
            }
        }
        _Float16* padp = bufA + m * SA + 36 + half * 14;
        #pragma unroll
        for (int c = 0; c < 14; ++c) padp[c] = (_Float16)0.0f;

        // ---- direction encoding stash (K = 24) ----
        float dv[3] = { directions[p * 3 + 0], directions[p * 3 + 1], directions[p * 3 + 2] };
        _Float16* dencp = dirS + m * 24 + half * 4;
        #pragma unroll
        for (int d = 0; d < 3; ++d) {
            float x = dv[d] * PI;
            float f = 1.0f;
            #pragma unroll
            for (int fr = 0; fr < 4; ++fr) {
                float ang = x * f;
                float sv = __sinf(ang), cv = __cosf(ang);
                float v = half ? cv : sv;
                dencp[d * 8 + fr] = (_Float16)v;
                f *= 2.0f;
            }
        }

        // ---- MLP trunk via WMMA ----
        layer<2, 8, true , SA, SB, G1_STRIDE>(bufA, sm.g1t, sm.gb[0], bufB, lane);
        layer<4, 8, true , SB, SA, GW_STRIDE>(bufB, sm.g2t, sm.gb[1], bufA, lane);
        layer<4, 8, true , SA, SB, GW_STRIDE>(bufA, sm.g3t, sm.gb[2], bufB, lane);
        layer<4, 8, true , SB, SA, GW_STRIDE>(bufB, sm.g4t, sm.gb[3], bufA, lane); // geo in A

        // ---- density head: softplus(geo . d_w + d_b), vectorized reads ----
        lds_fence();
        if (lane < 16) {
            float acc = sm.db[0];
            #pragma unroll
            for (int k = 0; k < 128; k += 8) {
                v8h h = *(const v8h*)(bufA + m * SA + k);
                #pragma unroll
                for (int i = 0; i < 8; ++i) acc += (float)h[i] * sm.dw[k + i];
            }
            float sp = (acc > 20.0f) ? acc : log1pf(__expf(acc));
            out[3 * NPTS + p] = sp;
        }

        // ---- feature (linear) ----
        layer<4, 8, false, SA, SB, GW_STRIDE>(bufA, sm.ft, sm.gb[4], bufB, lane);

        // ---- append dir encoding: color_in = [feature(128), dir(24), pad(8)] ----
        {
            _Float16* dst = bufB + m * SB + 128 + half * 12;
            const _Float16* src = dirS + m * 24 + half * 12;
            #pragma unroll
            for (int j = 0; j < 12; ++j) dst[j] = src[j];
            _Float16* zp = bufB + m * SB + 152 + half * 4;
            #pragma unroll
            for (int c = 0; c < 4; ++c) zp[c] = (_Float16)0.0f;
        }

        // ---- color layer 1 (152->64, relu), K padded to 160 ----
        layer<5, 4, true , SB, SA, C1_STRIDE>(bufB, sm.c1t, sm.c1b, bufA, lane);

        // ---- color head: sigmoid(c @ c2_w + c2_b), vectorized reads ----
        lds_fence();
        if (lane < 16) {
            float r0 = sm.c2b[0], r1 = sm.c2b[1], r2 = sm.c2b[2];
            #pragma unroll
            for (int k = 0; k < 64; k += 8) {
                v8h h = *(const v8h*)(bufA + m * SA + k);
                #pragma unroll
                for (int i = 0; i < 8; ++i) {
                    float a = (float)h[i];
                    r0 += a * sm.c2w[(k + i) * 3 + 0];
                    r1 += a * sm.c2w[(k + i) * 3 + 1];
                    r2 += a * sm.c2w[(k + i) * 3 + 2];
                }
            }
            out[p * 3 + 0] = 1.0f / (1.0f + __expf(-r0));
            out[p * 3 + 1] = 1.0f / (1.0f + __expf(-r1));
            out[p * 3 + 2] = 1.0f / (1.0f + __expf(-r2));
        }
    }
}

extern "C" void kernel_launch(void* const* d_in, const int* in_sizes, int n_in,
                              void* d_out, int out_size, void* d_ws, size_t ws_size,
                              hipStream_t stream) {
    (void)in_sizes; (void)n_in; (void)out_size; (void)d_ws; (void)ws_size;
    const float* positions  = (const float*)d_in[0];
    const float* directions = (const float*)d_in[1];
    const float* g1_w = (const float*)d_in[2];  const float* g1_b = (const float*)d_in[3];
    const float* g2_w = (const float*)d_in[4];  const float* g2_b = (const float*)d_in[5];
    const float* g3_w = (const float*)d_in[6];  const float* g3_b = (const float*)d_in[7];
    const float* g4_w = (const float*)d_in[8];  const float* g4_b = (const float*)d_in[9];
    const float* d_w  = (const float*)d_in[10]; const float* d_b  = (const float*)d_in[11];
    const float* f_w  = (const float*)d_in[12]; const float* f_b  = (const float*)d_in[13];
    const float* c1_w = (const float*)d_in[14]; const float* c1_b = (const float*)d_in[15];
    const float* c2_w = (const float*)d_in[16]; const float* c2_b = (const float*)d_in[17];
    float* out = (float*)d_out;

    tinynerf_wmma<<<GRID_BLOCKS, BLOCK_THREADS, 0, stream>>>(
        positions, directions, g1_w, g1_b, g2_w, g2_b, g3_w, g3_b, g4_w, g4_b,
        d_w, d_b, f_w, f_b, c1_w, c1_b, c2_w, c2_b, out);
}
